// SampleGrid2d_51445118272126
// MI455X (gfx1250) — compile-verified, matrix operand order checked by reading, not run on hardware
//
#include <hip/hip_runtime.h>
#include <hip/hip_bf16.h>

typedef __attribute__((ext_vector_type(16))) _Float16 v16h;
typedef __attribute__((ext_vector_type(8)))  _Float16 v8h;
typedef __attribute__((ext_vector_type(8)))  float    v8f;
typedef __attribute__((ext_vector_type(4)))  float    v4f;

#define B_N    4
#define CIN    256
#define COUT   64
#define H2D    224
#define W2D    224
#define HW     (H2D * W2D)      // 50176
#define GRID3  262144           // 64*64*64
#define LDSW_STRIDE 264         // f16 units per conv_w row in LDS (256 + 8 pad)

// ---------------------------------------------------------------------------
// Kernel 1: fused 1x1 conv as WMMA GEMM.
//   compT[b][hw][o] = sum_c conv_w[o][c] * grid2d[b][c][hw] + conv_b[o]
// Block = 256 threads = 8 waves. Each wave: one 16-pixel tile x all 64 outputs.
// A (16x32 f16) from grid2d (coalesced across half-wave), B (32x16 f16) from
// LDS-staged f16 conv_w, C/D f32 accumulators initialized with bias.
// ---------------------------------------------------------------------------
__global__ __launch_bounds__(256) void conv1x1_wmma_kernel(
    const float* __restrict__ grid2d,   // [B][CIN][HW]
    const float* __restrict__ conv_w,   // [COUT][CIN]
    const float* __restrict__ conv_b,   // [COUT]
    float* __restrict__ compT)          // [B][HW][COUT]
{
    __shared__ __align__(16) _Float16 ldsW[COUT * LDSW_STRIDE];

    const int b   = blockIdx.y;
    const int tid = threadIdx.x;

    // Stage conv_w into LDS as f16, row-major [o][c] with padded stride.
    for (int i = tid; i < (COUT * CIN) / 4; i += 256) {
        const int o = (i * 4) / CIN;
        const int c = (i * 4) % CIN;
        v4f w = *(const v4f*)(conv_w + (size_t)o * CIN + c);
        _Float16* dst = &ldsW[o * LDSW_STRIDE + c];
        dst[0] = (_Float16)w.x; dst[1] = (_Float16)w.y;
        dst[2] = (_Float16)w.z; dst[3] = (_Float16)w.w;
    }
    __syncthreads();

    const int wave = tid >> 5;
    const int lane = tid & 31;
    const int half = lane >> 4;          // 0: K-group base +0, 1: base +8
    const int l16  = lane & 15;

    const int hwtile = (blockIdx.x * 8 + wave) * 16;   // 50176/16 = 3136 tiles
    const int hw     = hwtile + l16;

    const float* gbase = grid2d + (size_t)b * CIN * HW + hw;

    v8f acc[4];
    #pragma unroll
    for (int t = 0; t < 4; ++t) {
        const float bias = conv_b[t * 16 + l16];   // lane's N column = output o
        #pragma unroll
        for (int r = 0; r < 8; ++r) acc[t][r] = bias;
    }

    for (int kk = 0; kk < CIN; kk += 32) {
        const int cb = kk + half * 8;

        // A fragment: lane = pixel row, 16 f16 covering c = cb..cb+7, cb+16..cb+23.
        v16h a;
        #pragma unroll
        for (int j = 0; j < 8; ++j)
            a[j] = (_Float16)gbase[(size_t)(cb + j) * HW];
        #pragma unroll
        for (int j = 0; j < 8; ++j)
            a[8 + j] = (_Float16)gbase[(size_t)(cb + 16 + j) * HW];

        #pragma unroll
        for (int t = 0; t < 4; ++t) {
            const int o = t * 16 + l16;
            const _Float16* wrow = &ldsW[o * LDSW_STRIDE + cb];
            v8h lo = *(const v8h*)(wrow);        // c = cb..cb+7     (16B aligned)
            v8h hi = *(const v8h*)(wrow + 16);   // c = cb+16..cb+23 (16B aligned)
            v16h bf;
            #pragma unroll
            for (int j = 0; j < 8; ++j) { bf[j] = lo[j]; bf[8 + j] = hi[j]; }

            acc[t] = __builtin_amdgcn_wmma_f32_16x16x32_f16(
                false, a, false, bf, (short)0, acc[t], false, false);
        }
    }

    // Store: lane l holds o = t*16 + l16; VGPR r holds pixel row r + 8*half.
    float* out = compT + ((size_t)b * HW + hwtile) * COUT;
    #pragma unroll
    for (int t = 0; t < 4; ++t) {
        const int o = t * 16 + l16;
        #pragma unroll
        for (int r = 0; r < 8; ++r) {
            const int m = r + half * 8;
            out[(size_t)m * COUT + o] = acc[t][r];
        }
    }
}

// ---------------------------------------------------------------------------
// Kernel 2: voxel projection + gather.
// One thread per voxel; 64-channel contiguous 256B read (L2-resident compT),
// coalesced strided writes to out[b][o][d][h][w].
// ---------------------------------------------------------------------------
__global__ __launch_bounds__(256) void gather_voxels_kernel(
    const float* __restrict__ compT,    // [B][HW][COUT]
    const float* __restrict__ projm,    // [B][4][4]
    const float* __restrict__ loc,      // [B][3]
    float* __restrict__ out)            // [B][COUT][64][64][64]
{
    const int b = blockIdx.y;
    const int v = blockIdx.x * 256 + threadIdx.x;  // 0..262143 (w fastest)
    const int w = v & 63;
    const int h = (v >> 6) & 63;
    const int d = v >> 12;

    const float* M = projm + b * 16;
    const float cx = (float)w + loc[b * 3 + 0];
    const float cy = (float)h + loc[b * 3 + 1];
    const float cz = (float)d + loc[b * 3 + 2];

    const float p0 = M[0]  * cx + M[1]  * cy + M[2]  * cz + M[3];
    const float p1 = M[4]  * cx + M[5]  * cy + M[6]  * cz + M[7];
    const float p2 = M[8]  * cx + M[9]  * cy + M[10] * cz + M[11];  // depth
    const float p3 = M[12] * cx + M[13] * cy + M[14] * cz + M[15];

    const float x = (p0 / p3) * 0.5f + 0.5f;
    const float y = (p1 / p3) * 0.5f + 0.5f;
    const int px = (int)(x * (float)W2D);   // truncation; v_cvt saturates like XLA
    const int py = (int)(y * (float)H2D);
    const int xx = min(max(px + 1, 0), W2D + 1);
    const int yy = min(max(py + 1, 0), H2D + 1);

    const bool inb = (xx >= 1) && (xx <= W2D) && (yy >= 1) && (yy <= H2D) &&
                     (p2 >= 0.0f);

    v4f vals[16];
    if (inb) {
        const size_t pix = (size_t)(yy - 1) * W2D + (size_t)(xx - 1);
        const float* src = compT + ((size_t)b * HW + pix) * COUT;
        #pragma unroll
        for (int i = 0; i < 16; ++i)
            vals[i] = *(const v4f*)(src + 4 * i);
    } else {
        #pragma unroll
        for (int i = 0; i < 16; ++i) {
            vals[i].x = 0.0f; vals[i].y = 0.0f; vals[i].z = 0.0f; vals[i].w = 0.0f;
        }
    }

    float* ob = out + (size_t)b * COUT * GRID3 + v;
    #pragma unroll
    for (int i = 0; i < 16; ++i) {
        ob[(size_t)(4 * i + 0) * GRID3] = vals[i].x;
        ob[(size_t)(4 * i + 1) * GRID3] = vals[i].y;
        ob[(size_t)(4 * i + 2) * GRID3] = vals[i].z;
        ob[(size_t)(4 * i + 3) * GRID3] = vals[i].w;
    }
}

extern "C" void kernel_launch(void* const* d_in, const int* in_sizes, int n_in,
                              void* d_out, int out_size, void* d_ws, size_t ws_size,
                              hipStream_t stream) {
    const float* grid2d = (const float*)d_in[0];   // (4,256,224,224)
    const float* projm  = (const float*)d_in[1];   // (4,4,4)
    const float* loc    = (const float*)d_in[2];   // (4,3)
    const float* conv_w = (const float*)d_in[3];   // (64,256)
    const float* conv_b = (const float*)d_in[4];   // (64,)

    float* compT = (float*)d_ws;                   // [4][50176][64] f32 = 51.4 MB
    float* outp  = (float*)d_out;                  // (4,64,64,64,64)

    // GEMM: 3136 pixel tiles per batch / 8 waves per block = 392 blocks x 4 batches
    dim3 gemm_grid(HW / (16 * 8), B_N);
    conv1x1_wmma_kernel<<<gemm_grid, 256, 0, stream>>>(grid2d, conv_w, conv_b, compT);

    // Gather: 262144 voxels per batch / 256 threads = 1024 blocks x 4 batches
    dim3 gat_grid(GRID3 / 256, B_N);
    gather_voxels_kernel<<<gat_grid, 256, 0, stream>>>(compT, projm, loc, outp);
}
